// SelfModifyingDeltaMemory_4303557230749
// MI455X (gfx1250) — matrix operand order repaired and checked by reading, not compiled, not run on hardware
//
#include <hip/hip_runtime.h>

typedef __attribute__((ext_vector_type(16))) _Float16 v16h;
typedef __attribute__((ext_vector_type(8)))  _Float16 v8h;
typedef __attribute__((ext_vector_type(4)))  _Float16 v4h;
typedef __attribute__((ext_vector_type(8)))  float    v8f;

#define DD   128
#define SS   256
#define BBN  16
#define ROWS (BBN * SS)   // 4096

#if defined(__HIP_DEVICE_COMPILE__) && __has_builtin(__builtin_amdgcn_global_load_async_to_lds_b128)
#define HAVE_ASYNC_LDS 1
// match the builtin's parameter types exactly (clang: "vector_size(16) int __device__*")
typedef int vint4 __attribute__((__vector_size__(16)));
typedef __attribute__((address_space(1))) vint4 g_vint4;   // global (__device__)
typedef __attribute__((address_space(3))) vint4 l_vint4;   // LDS (__shared__)
#endif

__device__ __forceinline__ float sigmoidf_(float x) { return 1.0f / (1.0f + __expf(-x)); }
__device__ __forceinline__ float siluf_(float x)    { return x * sigmoidf_(x); }

// ---------------------------------------------------------------------------
// 16x16 output tile of (A[rows,128] f16) x (Bt[cols,128] f16, pre-transposed)
// accumulated in f32 via v_wmma_f32_16x16x32_f16 (K swept 0..128 in 4 steps).
// A-frag layout: lane l16 = row, elems = 16 contiguous K at kb+16*hi.
// B-frag: lane l16 = column, elems = 16
// contiguous K at kb+16*hi (weights stored [n][k]).
// ---------------------------------------------------------------------------
__device__ __forceinline__ v8f wmma_rowtile(const _Float16* __restrict__ A,
                                            const _Float16* __restrict__ Bt,
                                            int row0, int col0) {
  const int lane = threadIdx.x & 31;
  const int l16  = lane & 15;
  const int hi   = lane >> 4;
  const _Float16* arow = A  + (size_t)(row0 + l16) * DD;
  const _Float16* brow = Bt + (size_t)(col0 + l16) * DD;
  v8f c = {};
#pragma unroll
  for (int kb = 0; kb < DD; kb += 32) {
    v8h alo = *(const v8h*)(arow + kb + hi * 16);
    v8h ahi = *(const v8h*)(arow + kb + hi * 16 + 8);
    v16h a, b;
#pragma unroll
    for (int i = 0; i < 8; ++i) { a[i] = alo[i]; a[i + 8] = ahi[i]; }
    b = *(const v16h*)(brow + kb + hi * 16);
    c = __builtin_amdgcn_wmma_f32_16x16x32_f16(false, a, false, b, (short)0, c,
                                               false, false);
  }
  return c;
}

// ------------------------------- prep kernels ------------------------------
__global__ void k_f32_to_f16(const float* __restrict__ src, _Float16* __restrict__ dst, int n) {
  int i = (blockIdx.x * blockDim.x + threadIdx.x) * 4;
  if (i + 3 < n) {
    float4 v = *(const float4*)(src + i);
    v4h h; h[0] = (_Float16)v.x; h[1] = (_Float16)v.y; h[2] = (_Float16)v.z; h[3] = (_Float16)v.w;
    *(v4h*)(dst + i) = h;
  }
}

// W is [K][N] row-major f32; produce Wt[N][K] f16
__global__ void k_transpose_f16(const float* __restrict__ W, _Float16* __restrict__ Wt,
                                int K, int N) {
  int i = blockIdx.x * blockDim.x + threadIdx.x;
  if (i < K * N) {
    int k = i / N, n = i % N;
    Wt[(size_t)n * K + k] = (_Float16)W[i];
  }
}

// ------------------------------- projections -------------------------------
// blockIdx.y selects Wq / Wk / Wv.  8 waves = 8 column tiles of one 16-row tile.
__global__ void k_gemm_qkv(const _Float16* __restrict__ xh,
                           const _Float16* __restrict__ WqT, const _Float16* __restrict__ WkT,
                           const _Float16* __restrict__ WvT,
                           float* __restrict__ Q, float* __restrict__ Kb,
                           _Float16* __restrict__ Vh) {
  int rt = blockIdx.x * 16;
  int n0 = (threadIdx.x >> 5) * 16;
  int sel = blockIdx.y;
  const _Float16* Bt = sel == 0 ? WqT : (sel == 1 ? WkT : WvT);
  v8f c = wmma_rowtile(xh, Bt, rt, n0);
  int lane = threadIdx.x & 31, l16 = lane & 15, hi = lane >> 4;
  int col = n0 + l16;
  if (sel == 2) {
#pragma unroll
    for (int r = 0; r < 8; ++r) Vh[(size_t)(rt + r + 8 * hi) * DD + col] = (_Float16)c[r];
  } else {
    float* O = sel == 0 ? Q : Kb;
#pragma unroll
    for (int r = 0; r < 8; ++r) O[(size_t)(rt + r + 8 * hi) * DD + col] = c[r];
  }
}

// G = silu(A*Bt + bias), stored f16 (feeds next GEMM)
__global__ void k_gemm_silu_f16(const _Float16* __restrict__ A, const _Float16* __restrict__ Bt,
                                const float* __restrict__ bias, _Float16* __restrict__ O) {
  int rt = blockIdx.x * 16;
  int n0 = (threadIdx.x >> 5) * 16;
  v8f c = wmma_rowtile(A, Bt, rt, n0);
  int lane = threadIdx.x & 31, l16 = lane & 15, hi = lane >> 4;
  int col = n0 + l16;
  float b0 = bias[col];
#pragma unroll
  for (int r = 0; r < 8; ++r)
    O[(size_t)(rt + r + 8 * hi) * DD + col] = (_Float16)siluf_(c[r] + b0);
}

// Vhat = A*Bt + bias, f32
__global__ void k_gemm_bias_f32(const _Float16* __restrict__ A, const _Float16* __restrict__ Bt,
                                const float* __restrict__ bias, float* __restrict__ O) {
  int rt = blockIdx.x * 16;
  int n0 = (threadIdx.x >> 5) * 16;
  v8f c = wmma_rowtile(A, Bt, rt, n0);
  int lane = threadIdx.x & 31, l16 = lane & 15, hi = lane >> 4;
  int col = n0 + l16;
  float b0 = bias[col];
#pragma unroll
  for (int r = 0; r < 8; ++r)
    O[(size_t)(rt + r + 8 * hi) * DD + col] = c[r] + b0;
}

// eta / alpha head: 4 waves -> E1 tiles (cols 0-15,16-31) and A1 tiles, then
// per-row 32-dot with We2/Wa2 + sigmoid scaling.
__global__ void k_gemm_ea(const _Float16* __restrict__ xh,
                          const _Float16* __restrict__ We1T, const _Float16* __restrict__ Wa1T,
                          const float* __restrict__ be1, const float* __restrict__ ba1,
                          const float* __restrict__ We2, const float* __restrict__ be2,
                          const float* __restrict__ Wa2, const float* __restrict__ ba2,
                          float* __restrict__ eta, float* __restrict__ alpha) {
  __shared__ float sE[16][32];
  __shared__ float sA[16][32];
  int rt = blockIdx.x * 16;
  int w  = threadIdx.x >> 5;          // 0..3
  int n0 = (w & 1) * 16;
  bool isA = w >= 2;
  v8f c = wmma_rowtile(xh, isA ? Wa1T : We1T, rt, n0);
  int lane = threadIdx.x & 31, l16 = lane & 15, hi = lane >> 4;
  int col = n0 + l16;
  float b0 = (isA ? ba1 : be1)[col];
#pragma unroll
  for (int r = 0; r < 8; ++r) {
    float v = siluf_(c[r] + b0);
    if (isA) sA[r + 8 * hi][col] = v; else sE[r + 8 * hi][col] = v;
  }
  __syncthreads();
  int tid = threadIdx.x;
  if (tid < 16) {
    float s = be2[0];
    for (int j = 0; j < 32; ++j) s += sE[tid][j] * We2[j];
    eta[rt + tid] = sigmoidf_(s) * 0.1f + 0.01f;
  } else if (tid < 32) {
    int row = tid - 16;
    float s = ba2[0];
    for (int j = 0; j < 32; ++j) s += sA[row][j] * Wa2[j];
    alpha[rt + row] = sigmoidf_(s) * 0.5f + 0.5f;
  }
}

// row-wise k normalization (in place): one wave per row, 4 elems/lane
__global__ void k_knorm(float* __restrict__ Kb) {
  int row  = blockIdx.x * 8 + (threadIdx.x >> 5);
  int lane = threadIdx.x & 31;
  float v[4]; float s = 0.f;
#pragma unroll
  for (int i = 0; i < 4; ++i) { v[i] = Kb[(size_t)row * DD + lane + i * 32]; s += v[i] * v[i]; }
#pragma unroll
  for (int off = 16; off >= 1; off >>= 1) s += __shfl_xor(s, off, 32);
  float inv = 1.0f / fmaxf(sqrtf(s), 1e-12f);
#pragma unroll
  for (int i = 0; i < 4; ++i) Kb[(size_t)row * DD + lane + i * 32] = v[i] * inv;
}

// ------------------------------- serial scan -------------------------------
// One block per sequence. M[128][128] distributed: thread t owns row r=t>>1,
// 64 columns starting at (t&1)*64.  Chunks of 32 steps staged in LDS via the
// CDNA5 async global->LDS path (ASYNCcnt) when available.
__global__ void __launch_bounds__(256, 1)
k_scan(const float* __restrict__ Q, const float* __restrict__ Kn,
       const float* __restrict__ VHAT, const float* __restrict__ eta,
       const float* __restrict__ alpha, _Float16* __restrict__ outH) {
  __shared__ float sq[32][DD];
  __shared__ float sk[32][DD];
  __shared__ float sv[32][DD];
  __shared__ float se[32];
  __shared__ float sa[32];
  const int bn  = blockIdx.x;
  const int tid = threadIdx.x;
  const int r   = tid >> 1;
  const int cb  = (tid & 1) * 64;
  float m[64];
#pragma unroll
  for (int c = 0; c < 64; ++c) m[c] = 0.f;

  for (int cc = 0; cc < 8; ++cc) {
    __syncthreads();                       // previous chunk fully consumed
    size_t base = ((size_t)bn * SS + cc * 32) * DD;
    const float4* q4 = (const float4*)(Q    + base);
    const float4* k4 = (const float4*)(Kn   + base);
    const float4* v4 = (const float4*)(VHAT + base);
#ifdef HAVE_ASYNC_LDS
    // async DMA: global -> LDS, 16B per lane per op, tracked by ASYNCcnt
#pragma unroll
    for (int j = 0; j < 4; ++j) {
      int idx = tid + j * 256;
      __builtin_amdgcn_global_load_async_to_lds_b128(
          (g_vint4*)(q4 + idx), (l_vint4*)&((float4*)sq)[idx], 0, 0);
      __builtin_amdgcn_global_load_async_to_lds_b128(
          (g_vint4*)(k4 + idx), (l_vint4*)&((float4*)sk)[idx], 0, 0);
      __builtin_amdgcn_global_load_async_to_lds_b128(
          (g_vint4*)(v4 + idx), (l_vint4*)&((float4*)sv)[idx], 0, 0);
    }
#else
#pragma unroll
    for (int j = 0; j < 4; ++j) {
      int idx = tid + j * 256;
      ((float4*)sq)[idx] = q4[idx];
      ((float4*)sk)[idx] = k4[idx];
      ((float4*)sv)[idx] = v4[idx];
    }
#endif
    if (tid < 32) {
      se[tid] = eta[bn * SS + cc * 32 + tid];
      sa[tid] = alpha[bn * SS + cc * 32 + tid];
    }
    if (cc < 7) {                          // hint next chunk toward the caches
      __builtin_prefetch(Q    + base + 32 * DD, 0, 0);
      __builtin_prefetch(Kn   + base + 32 * DD, 0, 0);
      __builtin_prefetch(VHAT + base + 32 * DD, 0, 0);
    }
#ifdef HAVE_ASYNC_LDS
#if __has_builtin(__builtin_amdgcn_s_wait_asynccnt)
    __builtin_amdgcn_s_wait_asynccnt(0);
#else
    asm volatile("s_wait_asynccnt 0x0" ::: "memory");
#endif
#endif
    __syncthreads();

    for (int t = 0; t < 32; ++t) {
      const float* qr = &sq[t][cb];
      const float* kr = &sk[t][cb];
      float po = 0.f, pm = 0.f;
#pragma unroll
      for (int c = 0; c < 64; ++c) {
        float mv = m[c];
        po = fmaf(mv, qr[c], po);
        pm = fmaf(mv, kr[c], pm);
      }
      po += __shfl_xor(po, 1, 32);         // pair-reduce: both halves of row r
      pm += __shfl_xor(pm, 1, 32);
      if ((tid & 1) == 0)
        outH[((size_t)bn * SS + cc * 32 + t) * DD + r] = (_Float16)po;
      float d = se[t] * (sv[t][r] - pm);   // eta*(vhat - M k)
      float a = sa[t];
#pragma unroll
      for (int c = 0; c < 64; ++c) m[c] = fmaf(a, m[c], d * kr[c]);
    }
  }
}

// --------------------- output projection + residual + LN -------------------
__global__ void k_final(const _Float16* __restrict__ outH, const _Float16* __restrict__ WoT,
                        const float* __restrict__ bo, const float* __restrict__ x,
                        const float* __restrict__ gamma, const float* __restrict__ beta,
                        float* __restrict__ out) {
  __shared__ float sh[16][DD];
  __shared__ float smu[16], srs[16];
  int rt = blockIdx.x * 16;
  int n0 = (threadIdx.x >> 5) * 16;
  v8f c = wmma_rowtile(outH, WoT, rt, n0);
  int lane = threadIdx.x & 31, l16 = lane & 15, hi = lane >> 4;
  int col = n0 + l16;
  float b0 = bo[col];
#pragma unroll
  for (int r = 0; r < 8; ++r) {
    int rl = r + 8 * hi;
    sh[rl][col] = c[r] + b0 + x[(size_t)(rt + rl) * DD + col];
  }
  __syncthreads();
  int tid = threadIdx.x;
  if (tid < 16) {
    float s = 0.f, s2 = 0.f;
    for (int j = 0; j < DD; ++j) { float v = sh[tid][j]; s += v; s2 += v * v; }
    float mu  = s * (1.0f / DD);
    float var = s2 * (1.0f / DD) - mu * mu;
    smu[tid] = mu;
    srs[tid] = rsqrtf(var + 1e-5f);
  }
  __syncthreads();
#pragma unroll
  for (int j = 0; j < 8; ++j) {
    int li = tid + j * 256;
    int rl = li >> 7, cix = li & 127;
    out[(size_t)(rt + rl) * DD + cix] =
        (sh[rl][cix] - smu[rl]) * srs[rl] * gamma[cix] + beta[cix];
  }
}

// ------------------------------- launcher ----------------------------------
extern "C" void kernel_launch(void* const* d_in, const int* in_sizes, int n_in,
                              void* d_out, int out_size, void* d_ws, size_t ws_size,
                              hipStream_t stream) {
  (void)in_sizes; (void)n_in; (void)out_size; (void)ws_size;
  const float* x     = (const float*)d_in[0];
  const float* Wq    = (const float*)d_in[1];
  const float* Wk    = (const float*)d_in[2];
  const float* Wv    = (const float*)d_in[3];
  const float* Wg1   = (const float*)d_in[4];
  const float* bg1   = (const float*)d_in[5];
  const float* Wg2   = (const float*)d_in[6];
  const float* bg2   = (const float*)d_in[7];
  const float* We1   = (const float*)d_in[8];
  const float* be1   = (const float*)d_in[9];
  const float* We2   = (const float*)d_in[10];
  const float* be2   = (const float*)d_in[11];
  const float* Wa1   = (const float*)d_in[12];
  const float* ba1   = (const float*)d_in[13];
  const float* Wa2   = (const float*)d_in[14];
  const float* ba2   = (const float*)d_in[15];
  const float* Wo    = (const float*)d_in[16];
  const float* bo    = (const float*)d_in[17];
  const float* gamma = (const float*)d_in[18];
  const float* beta  = (const float*)d_in[19];

  char* ws = (char*)d_ws;
  size_t off = 0;
  auto alloc = [&](size_t bytes) -> void* {
    void* p = ws + off;
    off += (bytes + 255) & ~(size_t)255;
    return p;
  };
  _Float16* xh   = (_Float16*)alloc((size_t)ROWS * DD * 2);
  _Float16* WqT  = (_Float16*)alloc((size_t)DD * DD * 2);
  _Float16* WkT  = (_Float16*)alloc((size_t)DD * DD * 2);
  _Float16* WvT  = (_Float16*)alloc((size_t)DD * DD * 2);
  _Float16* Wg1T = (_Float16*)alloc((size_t)DD * DD * 2);
  _Float16* Wg2T = (_Float16*)alloc((size_t)DD * DD * 2);
  _Float16* WoT  = (_Float16*)alloc((size_t)DD * DD * 2);
  _Float16* We1T = (_Float16*)alloc((size_t)32 * DD * 2);
  _Float16* Wa1T = (_Float16*)alloc((size_t)32 * DD * 2);
  float*    Qf   = (float*)alloc((size_t)ROWS * DD * 4);
  float*    Kb   = (float*)alloc((size_t)ROWS * DD * 4);
  float*    VHAT = (float*)alloc((size_t)ROWS * DD * 4);
  _Float16* Vh   = (_Float16*)alloc((size_t)ROWS * DD * 2);
  _Float16* Gh   = (_Float16*)alloc((size_t)ROWS * DD * 2);
  _Float16* outH = (_Float16*)alloc((size_t)ROWS * DD * 2);
  float*    eta  = (float*)alloc((size_t)ROWS * 4);
  float*    alph = (float*)alloc((size_t)ROWS * 4);

  // prep: f16 input + transposed f16 weights
  k_f32_to_f16<<<ROWS * DD / 1024, 256, 0, stream>>>(x, xh, ROWS * DD);
  k_transpose_f16<<<64, 256, 0, stream>>>(Wq,  WqT,  DD, DD);
  k_transpose_f16<<<64, 256, 0, stream>>>(Wk,  WkT,  DD, DD);
  k_transpose_f16<<<64, 256, 0, stream>>>(Wv,  WvT,  DD, DD);
  k_transpose_f16<<<64, 256, 0, stream>>>(Wg1, Wg1T, DD, DD);
  k_transpose_f16<<<64, 256, 0, stream>>>(Wg2, Wg2T, DD, DD);
  k_transpose_f16<<<64, 256, 0, stream>>>(Wo,  WoT,  DD, DD);
  k_transpose_f16<<<16, 256, 0, stream>>>(We1, We1T, DD, 32);
  k_transpose_f16<<<16, 256, 0, stream>>>(Wa1, Wa1T, DD, 32);

  // projections (WMMA)
  k_gemm_qkv<<<dim3(ROWS / 16, 3), 256, 0, stream>>>(xh, WqT, WkT, WvT, Qf, Kb, Vh);
  k_knorm<<<ROWS / 8, 256, 0, stream>>>(Kb);
  k_gemm_silu_f16<<<ROWS / 16, 256, 0, stream>>>(Vh, Wg1T, bg1, Gh);
  k_gemm_bias_f32<<<ROWS / 16, 256, 0, stream>>>(Gh, Wg2T, bg2, VHAT);
  k_gemm_ea<<<ROWS / 16, 128, 0, stream>>>(xh, We1T, Wa1T, be1, ba1, We2, be2,
                                           Wa2, ba2, eta, alph);

  // serial delta-memory scan (register-resident 128x128 state per sequence)
  k_scan<<<BBN, 256, 0, stream>>>(Qf, Kb, VHAT, eta, alph, outH);

  // output projection + bias + residual + LayerNorm (WMMA + fused LN)
  k_final<<<ROWS / 16, 256, 0, stream>>>(outH, WoT, bo, x, gamma, beta, (float*)d_out);
}
